// MatchNet_57836029608083
// MI455X (gfx1250) — matrix operand order repaired and checked by reading, not compiled, not run on hardware
//
#include <hip/hip_runtime.h>
#include <math.h>

typedef __attribute__((ext_vector_type(16))) _Float16 v16h;
typedef __attribute__((ext_vector_type(8)))  _Float16 v8h;
typedef __attribute__((ext_vector_type(8)))  float    v8f;

#define LDT 40  // LDS row stride in f16 elements (80B = 20 banks, 16B aligned)

// ---------------- elementwise f32 -> f16 convert ----------------
__global__ void cvt_f16_kernel(const float* __restrict__ in,
                               _Float16* __restrict__ out, int n) {
  int i = blockIdx.x * blockDim.x + threadIdx.x;
  if (i < n) out[i] = (_Float16)in[i];
}

// ------------- tiled transpose + convert: out[C][R] = in[R][C] -------------
__global__ void transpose_cvt_kernel(const float* __restrict__ in,
                                     _Float16* __restrict__ out, int R, int C) {
  __shared__ float tile[16][17];
  int c = blockIdx.x * 16 + threadIdx.x;
  int r = blockIdx.y * 16 + threadIdx.y;
  tile[threadIdx.y][threadIdx.x] = in[(size_t)r * C + c];
  __syncthreads();
  int orow = blockIdx.x * 16 + threadIdx.y;  // = original column
  int ocol = blockIdx.y * 16 + threadIdx.x;  // = original row
  out[(size_t)orow * R + ocol] = (_Float16)tile[threadIdx.x][threadIdx.y];
}

// ---------------- sum(S*S) reduction -> scal[1] ----------------
__global__ void sumsq_kernel(const float* __restrict__ s,
                             float* __restrict__ scal, int n) {
  __shared__ float red[256];
  float acc = 0.0f;
  for (int i = blockIdx.x * 256 + threadIdx.x; i < n; i += gridDim.x * 256) {
    float v = s[i];
    acc += v * v;
  }
  red[threadIdx.x] = acc;
  __syncthreads();
  for (int off = 128; off > 0; off >>= 1) {
    if (threadIdx.x < off) red[threadIdx.x] += red[threadIdx.x + off];
    __syncthreads();
  }
  if (threadIdx.x == 0) atomicAdd(&scal[1], red[0]);
}

// scal[0] = tau = sigma = 0.9 / (sqrt(sum S*S) + 1e-6)
__global__ void tau_kernel(float* scal) {
  scal[0] = 0.9f / (sqrtf(scal[1]) + 1e-6f);
}

// ---- async global -> LDS 16B copy (CDNA5 GLOBAL_LOAD_ASYNC_TO_LDS_B128) ----
__device__ __forceinline__ void async_copy_b128(const _Float16* gsrc,
                                                _Float16* lds_dst) {
  const unsigned int ldsa =
      (unsigned int)(unsigned long long)(uintptr_t)lds_dst;  // LDS byte offset
  const unsigned long long ga = (unsigned long long)(uintptr_t)gsrc;
  asm volatile("global_load_async_to_lds_b128 %0, %1, off"
               :
               : "v"(ldsa), "v"(ga)
               : "memory");
}

__device__ __forceinline__ void wait_async0() {
  asm volatile("s_wait_asynccnt 0" ::: "memory");
}

// ---------------- WMMA GEMM, C = A[MxK] * Bt[NxK]^T, fused epilogues ------
// EPI 0: out16 = relu(acc + bias[n])                      (h1, h2; f16 out)
// EPI 1: out32 = relu(acc + bias[n])                      (b_rhs; f32 out)
// EPI 2: y = relu(y + tau*(acc - brhs)); write f32 + f16  (out32=y, aux=brhs)
// EPI 3: out32 = x - tau*(acc - w[n])                     (aux=x, bias=w_obj)
template <int EPI>
__global__ __launch_bounds__(256) void gemm_wmma_kernel(
    const _Float16* __restrict__ A, const _Float16* __restrict__ Bt,
    const float* __restrict__ bias, const float* __restrict__ scal,
    const float* __restrict__ aux, float* __restrict__ out32,
    _Float16* __restrict__ out16, int M, int N, int K) {
  // double-buffered tiles: 2 x (128 x 32) f16 for A and Bt
  __shared__ __align__(16) _Float16 As[2][128 * LDT];
  __shared__ __align__(16) _Float16 Bs[2][128 * LDT];

  const int tid   = threadIdx.x;
  const int lane  = tid & 31;
  const int wid   = tid >> 5;          // 8 waves
  const int wm    = (wid >> 1) * 32;   // wave row offset in 128x128 tile
  const int wn    = (wid & 1) * 64;    // wave col offset
  const int laneN = lane & 15;
  const int laneH = lane >> 4;
  const int bM = blockIdx.y * 128, bN = blockIdx.x * 128;

  // branchless slot decomposition: thread owns slots tid and tid+256
  const int row0 = tid >> 2,           k80 = (tid & 3) << 3;
  const int row1 = (tid + 256) >> 2,   k81 = k80;  // (tid+256)&3 == tid&3

  v8f acc[2][4] = {};

  const int nK = K >> 5;

  // ---- issue async fill of tile 0 into buffer 0 ----
  {
    const int k0 = 0;
    async_copy_b128(&A[(size_t)(bM + row0) * K + k0 + k80], &As[0][row0 * LDT + k80]);
    async_copy_b128(&Bt[(size_t)(bN + row0) * K + k0 + k80], &Bs[0][row0 * LDT + k80]);
    async_copy_b128(&A[(size_t)(bM + row1) * K + k0 + k81], &As[0][row1 * LDT + k81]);
    async_copy_b128(&Bt[(size_t)(bN + row1) * K + k0 + k81], &Bs[0][row1 * LDT + k81]);
  }
  wait_async0();
  __syncthreads();

  for (int kt = 0; kt < nK; ++kt) {
    const int cur = kt & 1;

    // ---- issue async fill of next tile into the other buffer ----
    if (kt + 1 < nK) {
      const int k0 = (kt + 1) << 5;
      const int nxt = cur ^ 1;
      async_copy_b128(&A[(size_t)(bM + row0) * K + k0 + k80], &As[nxt][row0 * LDT + k80]);
      async_copy_b128(&Bt[(size_t)(bN + row0) * K + k0 + k80], &Bs[nxt][row0 * LDT + k80]);
      async_copy_b128(&A[(size_t)(bM + row1) * K + k0 + k81], &As[nxt][row1 * LDT + k81]);
      async_copy_b128(&Bt[(size_t)(bN + row1) * K + k0 + k81], &Bs[nxt][row1 * LDT + k81]);
    }

    // ---- fragments per ISA VGPR layouts (wave32) ----
    // A 16x32: lane holds row (lane&15); K = {8*h..+7} and {16+8*h..+7}
    // B 32x16: lane holds col (lane&15); K = {8*h..+7} and {16+8*h..+7}
    const _Float16* Asb = As[cur];
    const _Float16* Bsb = Bs[cur];
    union Frag { v16h v; v8h h[2]; };
    Frag fa[2], fb[4];
    #pragma unroll
    for (int mt = 0; mt < 2; ++mt) {
      const int r = wm + mt * 16 + laneN;
      fa[mt].h[0] = *(const v8h*)&Asb[r * LDT + laneH * 8];
      fa[mt].h[1] = *(const v8h*)&Asb[r * LDT + 16 + laneH * 8];
    }
    #pragma unroll
    for (int nt = 0; nt < 4; ++nt) {
      const int c = wn + nt * 16 + laneN;
      fb[nt].h[0] = *(const v8h*)&Bsb[c * LDT + laneH * 8];
      fb[nt].h[1] = *(const v8h*)&Bsb[c * LDT + 16 + laneH * 8];
    }
    #pragma unroll
    for (int mt = 0; mt < 2; ++mt)
      #pragma unroll
      for (int nt = 0; nt < 4; ++nt)
        acc[mt][nt] = __builtin_amdgcn_wmma_f32_16x16x32_f16(
            false, fa[mt].v, false, fb[nt].v, (short)0, acc[mt][nt],
            false, false);

    // next buffer complete (own wave) + all waves done reading current buffer
    wait_async0();
    __syncthreads();
  }

  // Epilogue: D layout — VGPR e holds M = e + 8*(lane>=16), N = lane&15
  const float t = (EPI >= 2) ? scal[0] : 0.0f;
  #pragma unroll
  for (int mt = 0; mt < 2; ++mt) {
    #pragma unroll
    for (int nt = 0; nt < 4; ++nt) {
      const int c  = bN + wn + nt * 16 + laneN;
      const int r0 = bM + wm + mt * 16 + laneH * 8;
      #pragma unroll
      for (int e = 0; e < 8; ++e) {
        const size_t idx = (size_t)(r0 + e) * N + c;
        const float val = acc[mt][nt][e];
        if (EPI == 0) {
          out16[idx] = (_Float16)fmaxf(0.0f, val + bias[c]);
        } else if (EPI == 1) {
          out32[idx] = fmaxf(0.0f, val + bias[c]);
        } else if (EPI == 2) {
          const float yv = fmaxf(0.0f, out32[idx] + t * (val - aux[idx]));
          out32[idx] = yv;
          out16[idx] = (_Float16)yv;
        } else {
          out32[idx] = aux[idx] - t * (val - bias[c]);
        }
      }
    }
  }
}

// ------------- prox: per-row L2 prox + projection, emits x(f32), xb(f16) ---
__global__ __launch_bounds__(256) void prox_kernel(
    const float* __restrict__ v, const float* __restrict__ z,
    float* __restrict__ x, _Float16* __restrict__ xb16,
    const float* __restrict__ scal) {
  __shared__ float red[256];
  const int row = blockIdx.x;
  const int t   = threadIdx.x;
  const size_t base = (size_t)row * 512;
  const float tau = scal[0];

  const float d0 = v[base + t]       - z[base + t];
  const float d1 = v[base + t + 256] - z[base + t + 256];
  red[t] = d0 * d0 + d1 * d1;
  __syncthreads();
  for (int off = 128; off > 0; off >>= 1) {
    if (t < off) red[t] += red[t + off];
    __syncthreads();
  }
  const float nrm   = sqrtf(red[0]);
  const float scale = fmaxf(0.0f, 1.0f - tau / fmaxf(nrm, 1e-12f));

  {
    const float xo = x[base + t];
    const float xn = fmaxf(0.0f, z[base + t] + scale * d0);
    x[base + t]    = xn;
    xb16[base + t] = (_Float16)(2.0f * xn - xo);
  }
  {
    const float xo = x[base + t + 256];
    const float xn = fmaxf(0.0f, z[base + t + 256] + scale * d1);
    x[base + t + 256]    = xn;
    xb16[base + t + 256] = (_Float16)(2.0f * xn - xo);
  }
}

// ------------------------------ launcher ------------------------------
extern "C" void kernel_launch(void* const* d_in, const int* in_sizes, int n_in,
                              void* d_out, int out_size, void* d_ws,
                              size_t ws_size, hipStream_t stream) {
  const int B = 4096, D = 512, H = 2048, N = 512, KIT = 40;

  const float* X    = (const float*)d_in[0];  // [B, D] (z anchor too)
  const float* W1   = (const float*)d_in[1];  // [D, H]
  const float* b1   = (const float*)d_in[2];  // [H]
  const float* W2   = (const float*)d_in[3];  // [H, H]
  const float* b2   = (const float*)d_in[4];  // [H]
  const float* W3   = (const float*)d_in[5];  // [H, N]
  const float* b3   = (const float*)d_in[6];  // [N]
  const float* S    = (const float*)d_in[7];  // [D, N]
  const float* wobj = (const float*)d_in[8];  // [N]

  // ---- workspace carve-up ----
  char* p = (char*)d_ws;
  size_t off = 0;
  auto take = [&](size_t bytes) -> char* {
    char* q = p + off;
    off = (off + bytes + 255) & ~(size_t)255;
    return q;
  };
  _Float16* Xf16 = (_Float16*)take((size_t)B * D * 2);
  _Float16* W1t  = (_Float16*)take((size_t)H * D * 2);  // [N=H][K=D]
  _Float16* W2t  = (_Float16*)take((size_t)H * H * 2);
  _Float16* W3t  = (_Float16*)take((size_t)N * H * 2);  // [N][K=H]
  _Float16* S16  = (_Float16*)take((size_t)D * N * 2);  // row-major = Bt of S^T
  _Float16* St16 = (_Float16*)take((size_t)N * D * 2);  // transposed = Bt of S
  _Float16* h1   = (_Float16*)take((size_t)B * H * 2);
  _Float16* h2   = (_Float16*)take((size_t)B * H * 2);
  float*    brhs = (float*)take((size_t)B * N * 4);
  float*    xbuf = (float*)take((size_t)B * N * 4);
  float*    ybuf = (float*)take((size_t)B * N * 4);
  float*    vbuf = (float*)take((size_t)B * N * 4);
  _Float16* xb16 = (_Float16*)take((size_t)B * N * 2);
  _Float16* y16  = (_Float16*)take((size_t)B * N * 2);
  float*    scal = (float*)take(2 * sizeof(float));
  (void)ws_size; (void)in_sizes; (void)n_in; (void)out_size;

  // ---- zero-init state ----
  hipMemsetAsync(xbuf, 0, (size_t)B * N * 4, stream);
  hipMemsetAsync(ybuf, 0, (size_t)B * N * 4, stream);
  hipMemsetAsync(xb16, 0, (size_t)B * N * 2, stream);
  hipMemsetAsync(y16,  0, (size_t)B * N * 2, stream);
  hipMemsetAsync(scal, 0, 2 * sizeof(float), stream);

  // ---- precision conversion / layout prep ----
  cvt_f16_kernel<<<(B * D) / 256, 256, 0, stream>>>(X, Xf16, B * D);
  cvt_f16_kernel<<<(D * N) / 256, 256, 0, stream>>>(S, S16, D * N);
  transpose_cvt_kernel<<<dim3(H / 16, D / 16), dim3(16, 16), 0, stream>>>(W1, W1t, D, H);
  transpose_cvt_kernel<<<dim3(H / 16, H / 16), dim3(16, 16), 0, stream>>>(W2, W2t, H, H);
  transpose_cvt_kernel<<<dim3(N / 16, H / 16), dim3(16, 16), 0, stream>>>(W3, W3t, H, N);
  transpose_cvt_kernel<<<dim3(N / 16, D / 16), dim3(16, 16), 0, stream>>>(S, St16, D, N);

  // ---- tau = sigma = 0.9 / (||S||_F + 1e-6) ----
  sumsq_kernel<<<256, 256, 0, stream>>>(S, scal, D * N);
  tau_kernel<<<1, 1, 0, stream>>>(scal);

  // ---- MLP: b_rhs = relu(relu(relu(X W1+b1) W2+b2) W3+b3) ----
  gemm_wmma_kernel<0><<<dim3(H / 128, B / 128), 256, 0, stream>>>(
      Xf16, W1t, b1, scal, nullptr, nullptr, h1, B, H, D);
  gemm_wmma_kernel<0><<<dim3(H / 128, B / 128), 256, 0, stream>>>(
      h1, W2t, b2, scal, nullptr, nullptr, h2, B, H, H);
  gemm_wmma_kernel<1><<<dim3(N / 128, B / 128), 256, 0, stream>>>(
      h2, W3t, b3, scal, nullptr, brhs, nullptr, B, N, H);

  // ---- PDHG iterations ----
  for (int it = 0; it < KIT; ++it) {
    // y = relu(y + sigma*(xb @ S^T - b_rhs))
    gemm_wmma_kernel<2><<<dim3(N / 128, B / 128), 256, 0, stream>>>(
        xb16, S16, nullptr, scal, brhs, ybuf, y16, B, N, N);
    // v = x - tau*(y @ S - w_obj)
    gemm_wmma_kernel<3><<<dim3(N / 128, B / 128), 256, 0, stream>>>(
        y16, St16, wobj, scal, xbuf, vbuf, nullptr, B, N, N);
    // x = prox(v); xb = 2x - x_old
    prox_kernel<<<B, 256, 0, stream>>>(vbuf, X, xbuf, xb16, scal);
  }

  // ---- output ----
  hipMemcpyAsync(d_out, xbuf, (size_t)B * N * sizeof(float),
                 hipMemcpyDeviceToDevice, stream);
}
